// EvaluatorNetwork_87222195847433
// MI455X (gfx1250) — matrix operand order repaired and checked by reading, not compiled, not run on hardware
//
#include <hip/hip_runtime.h>
#include <hip/hip_bf16.h>
#include <math.h>

typedef __bf16 bf16_t;
typedef __bf16 v16bf __attribute__((ext_vector_type(16)));
typedef __bf16 v8bf  __attribute__((ext_vector_type(8)));
typedef __bf16 v4bf  __attribute__((ext_vector_type(4)));
typedef float  v8f   __attribute__((ext_vector_type(8)));

#define PI_F 3.14159265358979323846f
#define INV_SQRT_128 0.088388347648318447f

// ---------------------------------------------------------------------------
// Fragment loaders (wave32 WMMA; this file's contiguous k = 16*half + s order, same for A and B)
// A (16-bit 16x32): per lane two aligned 16B runs: K[kb+half*16..+8) and
// K[kb+half*16+8..+8)  -> caller passes row pointer already offset by half*16.
// B (32x16): per lane 16 consecutive K = one ci slice = 4 rows of 4 taps.
// ---------------------------------------------------------------------------
__device__ __forceinline__ v16bf load_a_frag(const bf16_t* __restrict__ ap) {
  v8bf lo = *(const v8bf*)ap;
  v8bf hi = *(const v8bf*)(ap + 8);
  return __builtin_shufflevector(lo, hi, 0, 1, 2, 3, 4, 5, 6, 7,
                                 8, 9, 10, 11, 12, 13, 14, 15);
}

__device__ __forceinline__ v16bf load_b_frag(const bf16_t* __restrict__ bp, int WiP) {
  v16bf b;
#pragma unroll
  for (int kh = 0; kh < 4; ++kh) {
    const bf16_t* rp = (const bf16_t*)__builtin_assume_aligned(bp + (size_t)kh * WiP, 4);
    v4bf v;
    __builtin_memcpy(&v, rp, 8);
    b[kh * 4 + 0] = v[0];
    b[kh * 4 + 1] = v[1];
    b[kh * 4 + 2] = v[2];
    b[kh * 4 + 3] = v[3];
  }
  return b;
}

// ---------------------------------------------------------------------------
// Stage 1: row[b,h,i] = (1/sqrt(128)) * sum_w (re + i*im)[b,h,w] * e^{-2pi i w i/128}
// (fft2 ortho followed by ifft along H == DFT along W only)
// ---------------------------------------------------------------------------
__global__ __launch_bounds__(256) void dft_rows_kernel(const float* __restrict__ img,
                                                       float* __restrict__ rowc) {
  int idx = blockIdx.x * 256 + threadIdx.x;     // over B*H*W = 16*128*128
  int i = idx & 127;
  int h = (idx >> 7) & 127;
  int b = idx >> 14;
  const float* re = img + (((size_t)b * 2 + 0) * 128 + h) * 128;
  const float* im = re + (size_t)128 * 128;
  float ang = -2.0f * PI_F * (float)i / 128.0f;
  float ss, sc;
  __sincosf(ang, &ss, &sc);
  float cr = 1.0f, cim = 0.0f, sr = 0.0f, si = 0.0f;
  for (int w = 0; w < 128; ++w) {
    float xr = re[w], xi = im[w];
    sr += xr * cr - xi * cim;
    si += xr * cim + xi * cr;
    float ncr = cr * sc - cim * ss;
    cim = cr * ss + cim * sc;
    cr = ncr;
  }
  rowc[(size_t)idx * 2 + 0] = sr * INV_SQRT_128;
  rowc[(size_t)idx * 2 + 1] = si * INV_SQRT_128;
}

// ---------------------------------------------------------------------------
// Stage 2: materialize the 262-channel spectral map, PADDED (130x130 planes,
// zero halo) so conv0's im2col gather needs no boundary branches.
// ---------------------------------------------------------------------------
__global__ __launch_bounds__(256) void build_x_kernel(const float* __restrict__ rowc,
                                                      const float* __restrict__ mask,
                                                      const float* __restrict__ memb,
                                                      bf16_t* __restrict__ X0,
                                                      long long total) {
  size_t idx = (size_t)blockIdx.x * 256 + threadIdx.x;   // over 16*262*130*130
  if (idx >= (size_t)total) return;
  int wp = (int)(idx % 130);
  size_t r0 = idx / 130;
  int hp = (int)(r0 % 130);
  size_t r1 = r0 / 130;
  int c = (int)(r1 % 262);
  int b = (int)(r1 / 262);
  int h = hp - 1, w = wp - 1;
  float val = 0.0f;
  if (h >= 0 && h < 128 && w >= 0 && w < 128) {
    if (c < 256) {
      int col = c & 127;
      size_t ri = (((size_t)b * 128 + h) * 128 + col) * 2;
      float rr = rowc[ri], rim = rowc[ri + 1];
      int k = (col * w) & 127;                       // exact mod-128 phase
      float ang = 2.0f * PI_F * (float)k / 128.0f;
      float ts, tc;
      __sincosf(ang, &ts, &tc);
      float twr = tc * INV_SQRT_128, twi = ts * INV_SQRT_128;
      val = (c < 128) ? (rr * twr - rim * twi) : (rr * twi + rim * twr);
      val += mask[b * 128 + col];
    } else {
      val = memb[(((size_t)b * 6 + (c - 256)) * 128 + h) * 128 + w];
    }
  }
  X0[idx] = (bf16_t)val;
}

// ---------------------------------------------------------------------------
// f32 -> bf16 weight conversion
// ---------------------------------------------------------------------------
__global__ __launch_bounds__(256) void f32_to_bf16_kernel(const float* __restrict__ src,
                                                          bf16_t* __restrict__ dst, int n) {
  int i = blockIdx.x * 256 + threadIdx.x;
  if (i < n) dst[i] = (bf16_t)src[i];
}

// ---------------------------------------------------------------------------
// Zero-fill (borders of padded activation buffers; interiors overwritten later)
// ---------------------------------------------------------------------------
__global__ __launch_bounds__(256) void fill_zero_bf16(bf16_t* __restrict__ p, int n) {
  int i = blockIdx.x * 256 + threadIdx.x;
  if (i < n) p[i] = (bf16_t)0.0f;
}

// ---------------------------------------------------------------------------
// Implicit-GEMM 4x4 stride-2 pad-1 conv, bf16 WMMA, f32 accumulate.
// Wave tile = 32(M) x 16(N) with two accumulators: per 32-K chunk the wave
// loads 2 A-fragments + 1 shared B-fragment and issues 2 back-to-back WMMAs
// (independent D registers, shared B -> no D->A/B hazard). 4 loads per WMMA
// instead of 6, and the scattered im2col gather traffic is halved.
// Software-pipelined (fragments for chunk i+1 load before WMMAs of chunk i);
// padded input -> branch-free; wave-uniform control flow -> EXEC all-ones.
// ---------------------------------------------------------------------------
__global__ __launch_bounds__(256) void conv4x4s2_wmma(
    const bf16_t* __restrict__ X, const bf16_t* __restrict__ Wt,
    const float* __restrict__ bias, bf16_t* __restrict__ outBf,
    float* __restrict__ outF32,
    int Ci, int Hi, int Wi, int Co, int Ho, int Wo, int ntiles, int fuseLeaky) {
  int tid = blockIdx.x * 256 + threadIdx.x;
  int wave = tid >> 5;
  int lane = tid & 31;
  int mtiles = Co >> 5;                 // 32 output channels per wave tile
  int tilesPB = mtiles * ntiles;
  int b = wave / tilesPB;
  int t = wave - b * tilesPB;
  int mt = t / ntiles;
  int nt = t - mt * ntiles;

  int half = lane >> 4;
  int m = lane & 15;
  int p = nt * 16 + (lane & 15);
  int oh = p / Wo, ow = p - oh * Wo;

  int WiP = Wi + 2;
  size_t planeP = (size_t)(Hi + 2) * WiP;
  size_t Ktot = (size_t)Ci * 16;

  const bf16_t* aptr0 = Wt + (size_t)(mt * 32 + m) * Ktot + half * 16;
  const bf16_t* aptr1 = aptr0 + 16 * Ktot;
  // top-left tap of this pixel's 4x4 window (row 2*oh, col 2*ow in padded
  // plane), in the ci slice owned by this lane group (ci = half, +2 per chunk)
  const bf16_t* bptr = X + ((size_t)b * Ci + half) * planeP +
                       (size_t)(2 * oh) * WiP + (2 * ow);
  const size_t bstep = 2 * planeP;

  int iters = (int)(Ktot >> 5);   // >= 1 always

  v8f acc0 = {}, acc1 = {};
  v16bf a0 = load_a_frag(aptr0);
  v16bf a1 = load_a_frag(aptr1);
  v16bf bf = load_b_frag(bptr, WiP);
#pragma unroll 2
  for (int it = 1; it < iters; ++it) {
    aptr0 += 32;
    aptr1 += 32;
    bptr += bstep;
    v16bf a0n = load_a_frag(aptr0);
    v16bf a1n = load_a_frag(aptr1);
    v16bf bn = load_b_frag(bptr, WiP);
    __builtin_prefetch((const void*)(aptr0 + 512), 0, 1);
    acc0 = __builtin_amdgcn_wmma_f32_16x16x32_bf16(false, a0, false, bf,
                                                   (short)0, acc0, false, false);
    acc1 = __builtin_amdgcn_wmma_f32_16x16x32_bf16(false, a1, false, bf,
                                                   (short)0, acc1, false, false);
    a0 = a0n;
    a1 = a1n;
    bf = bn;
  }
  acc0 = __builtin_amdgcn_wmma_f32_16x16x32_bf16(false, a0, false, bf,
                                                 (short)0, acc0, false, false);
  acc1 = __builtin_amdgcn_wmma_f32_16x16x32_bf16(false, a1, false, bf,
                                                 (short)0, acc1, false, false);

  // Epilogue. C/D layout: VGPR r -> M = r + 8*half, N = lane&15.
  size_t oPlaneP = (size_t)(Ho + 2) * (Wo + 2);
  size_t oPlane = (size_t)Ho * Wo;
#pragma unroll
  for (int r = 0; r < 8; ++r) {
#pragma unroll
    for (int g = 0; g < 2; ++g) {
      int co = mt * 32 + g * 16 + r + half * 8;
      float y = (g == 0 ? acc0[r] : acc1[r]) + bias[co];
      if (fuseLeaky) {
        // write into PADDED bf16 output (halo pre-zeroed)
        y = y < 0.0f ? 0.2f * y : y;
        size_t oidx = ((size_t)b * Co + co) * oPlaneP +
                      (size_t)(oh + 1) * (Wo + 2) + (ow + 1);
        outBf[oidx] = (bf16_t)y;
      } else {
        // raw f32, unpadded, for the InstanceNorm pass
        size_t oidx = ((size_t)b * Co + co) * oPlane + p;
        outF32[oidx] = y;
      }
    }
  }
}

// ---------------------------------------------------------------------------
// InstanceNorm (biased var, eps=1e-5) + LeakyReLU(0.2), f32 -> PADDED bf16.
// One 256-thread block per (b, channel) plane; LDS tree reduction.
// ---------------------------------------------------------------------------
__global__ __launch_bounds__(256) void inorm_leaky_kernel(const float* __restrict__ Z,
                                                          bf16_t* __restrict__ A,
                                                          int HW, int Wo, int planeP) {
  __shared__ float s1[256], s2[256];
  int plane = blockIdx.x;
  int t = threadIdx.x;
  const float* zp = Z + (size_t)plane * HW;
  float s = 0.0f, q = 0.0f;
  for (int i = t; i < HW; i += 256) {
    float v = zp[i];
    s += v;
    q += v * v;
  }
  s1[t] = s;
  s2[t] = q;
  __syncthreads();
  for (int off = 128; off > 0; off >>= 1) {
    if (t < off) {
      s1[t] += s1[t + off];
      s2[t] += s2[t + off];
    }
    __syncthreads();
  }
  float mean = s1[0] / (float)HW;
  float var = s2[0] / (float)HW - mean * mean;
  float inv = rsqrtf(var + 1e-5f);
  bf16_t* ap = A + (size_t)plane * planeP;
  int WoP = Wo + 2;
  for (int i = t; i < HW; i += 256) {
    float v = (zp[i] - mean) * inv;
    v = v < 0.0f ? 0.2f * v : v;
    int ohh = i / Wo, oww = i - ohh * Wo;
    ap[(size_t)(ohh + 1) * WoP + (oww + 1)] = (bf16_t)v;
  }
}

// ---------------------------------------------------------------------------
// Final stage InstanceNorm + LeakyReLU fused with 8x8 avg-pool.
// One 64-thread block per (b, channel); HW == 64.
// ---------------------------------------------------------------------------
__global__ __launch_bounds__(64) void inorm_pool_kernel(const float* __restrict__ Z,
                                                        float* __restrict__ pooled) {
  __shared__ float s1[64], s2[64];
  int plane = blockIdx.x;
  int t = threadIdx.x;
  float v = Z[(size_t)plane * 64 + t];
  s1[t] = v;
  s2[t] = v * v;
  __syncthreads();
  for (int off = 32; off > 0; off >>= 1) {
    if (t < off) {
      s1[t] += s1[t + off];
      s2[t] += s2[t + off];
    }
    __syncthreads();
  }
  float mean = s1[0] * (1.0f / 64.0f);
  float var = s2[0] * (1.0f / 64.0f) - mean * mean;
  float inv = rsqrtf(var + 1e-5f);
  float u = (v - mean) * inv;
  u = u < 0.0f ? 0.2f * u : u;
  __syncthreads();
  s1[t] = u;
  __syncthreads();
  for (int off = 32; off > 0; off >>= 1) {
    if (t < off) s1[t] += s1[t + off];
    __syncthreads();
  }
  if (t == 0) pooled[plane] = s1[0] * (1.0f / 64.0f);
}

// ---------------------------------------------------------------------------
// Head: out[b,j] = sum_c pooled[b,c] * w4[j,c] + b4[j]   (16x128x1024, f32)
// ---------------------------------------------------------------------------
__global__ __launch_bounds__(128) void head_kernel(const float* __restrict__ pooled,
                                                   const float* __restrict__ w4,
                                                   const float* __restrict__ b4,
                                                   float* __restrict__ out) {
  int b = blockIdx.x;
  int j = threadIdx.x;
  const float* pp = pooled + (size_t)b * 1024;
  const float* wp = w4 + (size_t)j * 1024;
  float s = 0.0f;
  for (int c = 0; c < 1024; ++c) s += pp[c] * wp[c];
  out[b * 128 + j] = s + b4[j];
}

// ---------------------------------------------------------------------------
extern "C" void kernel_launch(void* const* d_in, const int* in_sizes, int n_in,
                              void* d_out, int out_size, void* d_ws, size_t ws_size,
                              hipStream_t stream) {
  const float* input = (const float*)d_in[0];   // (16,2,128,128)
  const float* memb  = (const float*)d_in[1];   // (16,6,128,128)
  const float* mask  = (const float*)d_in[2];   // (16,1,1,128)
  const float* w0 = (const float*)d_in[3];
  const float* b0 = (const float*)d_in[4];
  const float* w1 = (const float*)d_in[5];
  const float* b1 = (const float*)d_in[6];
  const float* w2 = (const float*)d_in[7];
  const float* b2 = (const float*)d_in[8];
  const float* w3 = (const float*)d_in[9];
  const float* b3 = (const float*)d_in[10];
  const float* w4 = (const float*)d_in[11];
  const float* b4 = (const float*)d_in[12];
  float* out = (float*)d_out;                   // (16,128) f32

  char* ws = (char*)d_ws;
  size_t off = 0;
  auto give = [&](size_t bytes) {
    size_t o = off;
    off += (bytes + 255) & ~(size_t)255;
    return o;
  };
  // padded plane sizes
  const size_t pX0  = (size_t)130 * 130;  // conv0 input planes
  const size_t pA0  = (size_t)66 * 66;    // act0 planes (64x64 + halo)
  const size_t pA1  = (size_t)34 * 34;    // act1 planes (32x32 + halo)
  const size_t pA2  = (size_t)18 * 18;    // act2 planes (16x16 + halo)

  float*  rowc  = (float*)(ws + give((size_t)16 * 128 * 128 * 2 * 4));
  bf16_t* X0    = (bf16_t*)(ws + give((size_t)16 * 262 * pX0 * 2));
  bf16_t* w0b   = (bf16_t*)(ws + give((size_t)256 * 4192 * 2));
  bf16_t* w1b   = (bf16_t*)(ws + give((size_t)512 * 4096 * 2));
  bf16_t* w2b   = (bf16_t*)(ws + give((size_t)1024 * 8192 * 2));
  bf16_t* w3b   = (bf16_t*)(ws + give((size_t)1024 * 16384 * 2));
  bf16_t* act0  = (bf16_t*)(ws + give((size_t)16 * 256 * pA0 * 2));
  float*  Z1    = (float*)(ws + give((size_t)16 * 512 * 32 * 32 * 4));
  bf16_t* act1  = (bf16_t*)(ws + give((size_t)16 * 512 * pA1 * 2));
  float*  Z2    = (float*)(ws + give((size_t)16 * 1024 * 16 * 16 * 4));
  bf16_t* act2  = (bf16_t*)(ws + give((size_t)16 * 1024 * pA2 * 2));
  float*  Z3    = (float*)(ws + give((size_t)16 * 1024 * 8 * 8 * 4));
  float*  pooled = (float*)(ws + give((size_t)16 * 1024 * 4));
  (void)ws_size; (void)n_in; (void)in_sizes; (void)out_size;

  // Spectral map (writes full padded buffer incl. zero halo)
  dft_rows_kernel<<<(16 * 128 * 128) / 256, 256, 0, stream>>>(input, rowc);
  {
    long long total = (long long)16 * 262 * (long long)pX0;
    unsigned blocks = (unsigned)((total + 255) / 256);
    build_x_kernel<<<blocks, 256, 0, stream>>>(rowc, mask, memb, X0, total);
  }

  // Weight conversion to bf16
  f32_to_bf16_kernel<<<(256 * 4192 + 255) / 256, 256, 0, stream>>>(w0, w0b, 256 * 4192);
  f32_to_bf16_kernel<<<(512 * 4096 + 255) / 256, 256, 0, stream>>>(w1, w1b, 512 * 4096);
  f32_to_bf16_kernel<<<(1024 * 8192 + 255) / 256, 256, 0, stream>>>(w2, w2b, 1024 * 8192);
  f32_to_bf16_kernel<<<(1024 * 16384 + 255) / 256, 256, 0, stream>>>(w3, w3b, 1024 * 16384);

  // Zero the padded activation buffers (halo regions; interiors overwritten)
  {
    int n0 = 16 * 256 * (int)pA0;
    fill_zero_bf16<<<(n0 + 255) / 256, 256, 0, stream>>>(act0, n0);
    int n1 = 16 * 512 * (int)pA1;
    fill_zero_bf16<<<(n1 + 255) / 256, 256, 0, stream>>>(act1, n1);
    int n2 = 16 * 1024 * (int)pA2;
    fill_zero_bf16<<<(n2 + 255) / 256, 256, 0, stream>>>(act2, n2);
  }

  // Wave tiles are 32(M) x 16(N): blocks = B * (Co/32) * (HoWo/16) / 8
  // conv0: 262x128x128 -> 256x64x64, fused bias+leaky -> padded bf16
  conv4x4s2_wmma<<<4096, 256, 0, stream>>>(X0, w0b, b0, act0, nullptr,
                                           262, 128, 128, 256, 64, 64, 256, 1);
  // conv1: 256x64x64 -> 512x32x32 (raw f32), then inorm+leaky -> padded bf16
  conv4x4s2_wmma<<<2048, 256, 0, stream>>>(act0, w1b, b1, nullptr, Z1,
                                           256, 64, 64, 512, 32, 32, 64, 0);
  inorm_leaky_kernel<<<16 * 512, 256, 0, stream>>>(Z1, act1, 1024, 32, (int)pA1);
  // conv2: 512x32x32 -> 1024x16x16
  conv4x4s2_wmma<<<1024, 256, 0, stream>>>(act1, w2b, b2, nullptr, Z2,
                                           512, 32, 32, 1024, 16, 16, 16, 0);
  inorm_leaky_kernel<<<16 * 1024, 256, 0, stream>>>(Z2, act2, 256, 16, (int)pA2);
  // conv3: 1024x16x16 -> 1024x8x8
  conv4x4s2_wmma<<<256, 256, 0, stream>>>(act2, w3b, b3, nullptr, Z3,
                                          1024, 16, 16, 1024, 8, 8, 4, 0);
  // inorm + leaky + 8x8 avg pool
  inorm_pool_kernel<<<16 * 1024, 64, 0, stream>>>(Z3, pooled);
  // head
  head_kernel<<<16, 128, 0, stream>>>(pooled, w4, b4, out);
}
